// EqvLEPFeedForward_51994874085875
// MI455X (gfx1250) — matrix operand order repaired and checked by reading, not compiled, not run on hardware
//
#include <hip/hip_runtime.h>
#include <hip/hip_bf16.h>
#include <hip/hip_fp16.h>
#include <math.h>

typedef __attribute__((ext_vector_type(16))) _Float16 v16h;
typedef __attribute__((ext_vector_type(2)))  __fp16   v2k;   // cvt_pkrtz result type
typedef __attribute__((ext_vector_type(8)))  float    v8f;
typedef __attribute__((ext_vector_type(2)))  float    v2f;

#define Z   4
#define N1  192
#define NN  384
#define CC  16
#define NB  10
#define HH  12
#define NTILE   24          // 384/16 a-tiles per z
#define KSPLIT  4           // K (b) range split across blocks
#define CHUNKS  (NN / 2)    // 192 WMMA K-chunks total
#define L2E     1.4426950408889634f

// Y0/sqrt(H) folded into G; (1/sqrt(NB))*(1/GAUSS_NORM) folded into W_rad1 in LDS.
#define SCALE_G   (0.28209479177387814f * 0.2886751345948129f)   // Y0 / sqrt(12)
#define WSCALE    (0.31622776601683794f * (1.0f / 1.423085244900308f))
#define CSTEP     (10.0f / 9.0f)                                  // linspace(0,10,10) step
#define IS2L2E    (1.265625f * L2E)                               // (1/sigma^2)*log2(e)

// ---------------------------------------------------------------------------
// Kernel 1: G16[z][b][i][h16] = (Y0/sqrt(H)) * sum_j W_rad_out[h, i*16+j] * x[z,b,j]
// ---------------------------------------------------------------------------
__global__ void prep_G_kernel(const float* __restrict__ input1,
                              const float* __restrict__ input2,
                              const float* __restrict__ W_rad_out,
                              _Float16* __restrict__ G16) {
    int idx = blockIdx.x * blockDim.x + threadIdx.x;   // z*N*256 + b*256 + i*16 + h
    int h = idx & 15;
    int i = (idx >> 4) & 15;
    int b = (idx >> 8) % NN;
    int z = idx / (NN * 256);
    const float* x = (b < N1) ? &input1[((size_t)z * N1 + b) * CC]
                              : &input2[((size_t)z * N1 + (b - N1)) * CC];
    float g = 0.0f;
    if (h < HH) {
        const float* w = &W_rad_out[(size_t)h * (CC * CC) + i * CC];
        float acc = 0.0f;
        #pragma unroll
        for (int j = 0; j < CC; ++j) acc += x[j] * w[j];
        g = acc * SCALE_G;
    }
    G16[idx] = (_Float16)g;
}

// ---------------------------------------------------------------------------
// Kernel 2: per (z, a-tile, K-slice): partial y[16x16] over 48 K-chunks via
// v_wmma_f32_16x16x32_f16.  Radial MLP fully in packed f32 (v_pk_fma_f32,
// v_exp_f32, v_rcp_f32) with a single v_cvt_pkrtz per A-fragment h-pair.
// B fragments stream through a running pointer; unconditional speculative
// prefetch of the next chunk's row (branch-free inner loop).
// ---------------------------------------------------------------------------
__global__ __launch_bounds__(256)
void pair_wmma_kernel(const float* __restrict__ xyz1,
                      const float* __restrict__ xyz2,
                      const float* __restrict__ W_rad1,
                      const _Float16* __restrict__ G16,
                      float* __restrict__ ypart) {
    __shared__ v2f  lds_w1p[NB * 8];    // [k][h-pair] f32 pairs, h>=12 zeroed
    __shared__ float lds_px[NN], lds_py[NN], lds_pz[NN];

    const int z   = blockIdx.y;
    const int tid = threadIdx.x;

    for (int t = tid; t < NB * 8; t += 256) {
        const int k = t >> 3, hp = t & 7;
        const int h0 = 2 * hp, h1 = h0 + 1;
        const float w0 = (h0 < HH) ? W_rad1[k * HH + h0] * WSCALE : 0.0f;
        const float w1 = (h1 < HH) ? W_rad1[k * HH + h1] * WSCALE : 0.0f;
        lds_w1p[t] = (v2f){w0, w1};
    }
    for (int t = tid; t < NN; t += 256) {
        const float* p = (t < N1) ? &xyz1[((size_t)z * N1 + t) * 3]
                                  : &xyz2[((size_t)z * N1 + (t - N1)) * 3];
        lds_px[t] = p[0]; lds_py[t] = p[1]; lds_pz[t] = p[2];
    }
    __syncthreads();

    const int wave = tid >> 5;
    const int lane = tid & 31;
    const int hl   = lane >> 4;         // lane half: 0 -> b0, 1 -> b0+1 (h 0..15 each)
    const int m    = lane & 15;         // matrix row (A) / column (B,C,D)
    const int tile = blockIdx.x * 8 + wave;   // 0..23
    const int a0   = tile * 16;

    const float ax = lds_px[a0 + m];
    const float ay = lds_py[a0 + m];
    const float az = lds_pz[a0 + m];

    v8f acc = {};

    const int chunk0 = (int)blockIdx.z * (CHUNKS / KSPLIT);
    // Running B-fragment pointer: row G16[z][b0+hl][m][0..15]; +512 elems per chunk.
    const _Float16* gp = G16 + (size_t)z * NN * 256
                             + (((size_t)(chunk0 * 2 + hl) * 16 + m) << 4);

    for (int chunk = chunk0; chunk < chunk0 + CHUNKS / KSPLIT; ++chunk, gp += 512) {
        const int b0 = chunk * 2;

        // B fragment: contiguous 32B row -> 2x global_load_b128
        v16h bfrag = *(const v16h*)gp;
        // Speculative prefetch of next chunk's row (harmless past the end).
        __builtin_prefetch((const void*)(gp + 512), 0, 1);

        // distances to both b's, packed f32
        v2f dxv = {ax - lds_px[b0], ax - lds_px[b0 + 1]};
        v2f dyv = {ay - lds_py[b0], ay - lds_py[b0 + 1]};
        v2f dzv = {az - lds_pz[b0], az - lds_pz[b0 + 1]};
        v2f ddv = {1e-12f, 1e-12f};
        ddv = __builtin_elementwise_fma(dxv, dxv, ddv);
        ddv = __builtin_elementwise_fma(dyv, dyv, ddv);
        ddv = __builtin_elementwise_fma(dzv, dzv, ddv);
        v2f dv    = {__builtin_amdgcn_sqrtf(ddv.x), __builtin_amdgcn_sqrtf(ddv.y)};
        v2f negtv = ddv * (-IS2L2E);

        // Gaussian basis in log2 domain for both b's: exp2(d*Ak - t - Bk)
        v2f bkv[NB];
        #pragma unroll
        for (int k = 0; k < NB; ++k) {
            const float ck = CSTEP * (float)k;
            const float Ak = 2.0f * ck * IS2L2E;
            const float Bk = ck * ck * IS2L2E;
            v2f ev = __builtin_elementwise_fma(dv, (v2f){Ak, Ak}, negtv);
            ev = ev - (v2f){Bk, Bk};
            bkv[k] = (v2f){__builtin_amdgcn_exp2f(ev.x), __builtin_amdgcn_exp2f(ev.y)};
        }

        // radial MLP + silu, two h's at a time, all packed f32; pack to f16 once
        union { v16h v; v2k h2[8]; } au;
        #pragma unroll
        for (int t = 0; t < 2; ++t) {
            #pragma unroll
            for (int hpi = 0; hpi < 4; ++hpi) {
                v2f s2 = {0.0f, 0.0f};
                #pragma unroll
                for (int k = 0; k < NB; ++k) {
                    v2f bsp = {bkv[k][hl], bkv[k][hl]};
                    s2 = __builtin_elementwise_fma(bsp, lds_w1p[k * 8 + t * 4 + hpi], s2);
                }
                v2f arg = s2 * (-L2E);
                v2f e   = {__builtin_amdgcn_exp2f(arg.x), __builtin_amdgcn_exp2f(arg.y)};
                v2f den = e + 1.0f;
                v2f sig = {__builtin_amdgcn_rcpf(den.x), __builtin_amdgcn_rcpf(den.y)};
                v2f sil = s2 * sig;                          // silu pair
                au.h2[t * 4 + hpi] = __builtin_amdgcn_cvt_pkrtz(sil.x, sil.y);
            }
        }

        acc = __builtin_amdgcn_wmma_f32_16x16x32_f16(
                  /*neg_a=*/false, au.v, /*neg_b=*/false, bfrag,
                  /*c_mod=*/(short)0, acc, /*reuse_a=*/false, /*reuse_b=*/false);
    }

    // Store partial tile in fragment order: 8 contiguous floats per lane.
    float* yp = ypart + ((((size_t)z * NTILE + tile) * KSPLIT + blockIdx.z) << 8) + lane * 8;
    #pragma unroll
    for (int r = 0; r < 8; ++r) yp[r] = acc[r];
}

// ---------------------------------------------------------------------------
// Kernel 2b: sum K-slices, apply |.| + mask, reduce over a into s[z][i].
// ---------------------------------------------------------------------------
__global__ void reduce_kernel(const float* __restrict__ ypart,
                              const int*   __restrict__ mask,
                              float* __restrict__ s_acc) {
    int idx  = blockIdx.x * blockDim.x + threadIdx.x;  // z*NTILE*256 + tile*256 + e
    int e    = idx & 255;
    int tile = (idx >> 8) % NTILE;
    int z    = idx / (NTILE * 256);
    const float* yp = ypart + (((size_t)z * NTILE + tile) * KSPLIT << 8) + e;
    float sum = 0.0f;
    #pragma unroll
    for (int s = 0; s < KSPLIT; ++s) sum += yp[(size_t)s << 8];
    const int lane = e >> 3, r = e & 7;
    const int row  = ((lane >> 4) << 3) + r;      // C/D fragment -> matrix row
    const int col  = lane & 15;
    const int a    = tile * 16 + row;
    float v = __builtin_fabsf(sum);
    v = (mask[z * NN + a] == 0) ? 0.0f : v;
    atomicAdd(&s_acc[z * CC + col], v);
}

// ---------------------------------------------------------------------------
// Kernel 0: zero the s accumulator (workspace is poisoned by the harness).
// ---------------------------------------------------------------------------
__global__ void init_kernel(float* __restrict__ s_acc) {
    if (threadIdx.x < Z * CC) s_acc[threadIdx.x] = 0.0f;
}

// ---------------------------------------------------------------------------
// Kernel 3: normalize (ddof=1) -> fc3 + leaky_relu(0.01) -> fc2 -> sigmoid.
// ---------------------------------------------------------------------------
__global__ void finalize_kernel(const float* __restrict__ s_acc,
                                const float* __restrict__ W_fc3,
                                const float* __restrict__ W_fc2,
                                float* __restrict__ out) {
    const int z = threadIdx.x;
    if (z >= Z) return;
    float s[CC];
    float mean = 0.0f;
    #pragma unroll
    for (int i = 0; i < CC; ++i) { s[i] = s_acc[z * CC + i]; mean += s[i]; }
    mean *= (1.0f / CC);
    float var = 0.0f;
    #pragma unroll
    for (int i = 0; i < CC; ++i) { const float d = s[i] - mean; var += d * d; }
    var *= (1.0f / (CC - 1));                       // unbiased std (ddof=1)
    const float inv = 1.0f / (__builtin_amdgcn_sqrtf(var) + 1e-6f);
    float sn[CC];
    #pragma unroll
    for (int i = 0; i < CC; ++i) sn[i] = (s[i] - mean) * inv;
    float o = 0.0f;
    #pragma unroll
    for (int k = 0; k < CC; ++k) {
        float a = 0.0f;
        #pragma unroll
        for (int i = 0; i < CC; ++i) a += sn[i] * W_fc3[i * CC + k];
        a *= 0.25f;                                 // 1/sqrt(C)
        a = (a > 0.0f) ? a : 0.01f * a;             // leaky_relu
        o += a * W_fc2[k];
    }
    o *= 0.25f;                                     // 1/sqrt(C)
    out[z] = __builtin_amdgcn_rcpf(1.0f + __expf(-o));   // sigmoid, no IEEE div
}

// ---------------------------------------------------------------------------
extern "C" void kernel_launch(void* const* d_in, const int* in_sizes, int n_in,
                              void* d_out, int out_size, void* d_ws, size_t ws_size,
                              hipStream_t stream) {
    (void)in_sizes; (void)n_in; (void)out_size; (void)ws_size;

    const float* input1    = (const float*)d_in[0];
    const float* input2    = (const float*)d_in[1];
    const float* xyz1      = (const float*)d_in[2];
    const float* xyz2      = (const float*)d_in[3];
    const int*   mask      = (const int*)d_in[4];
    const float* W_rad1    = (const float*)d_in[5];
    const float* W_rad_out = (const float*)d_in[6];
    const float* W_fc3     = (const float*)d_in[7];
    const float* W_fc2     = (const float*)d_in[8];
    float*       out       = (float*)d_out;

    const size_t G16_BYTES   = (size_t)Z * NN * 256 * sizeof(_Float16);   // 768 KB
    const size_t YPART_BYTES = (size_t)Z * NTILE * KSPLIT * 256 * sizeof(float); // 384 KB
    _Float16* G16   = (_Float16*)d_ws;
    float*    ypart = (float*)((char*)d_ws + G16_BYTES);
    float*    s_acc = (float*)((char*)d_ws + G16_BYTES + YPART_BYTES);

    init_kernel<<<1, 64, 0, stream>>>(s_acc);
    prep_G_kernel<<<(Z * NN * 256) / 256, 256, 0, stream>>>(input1, input2, W_rad_out, G16);
    dim3 grid(3, Z, KSPLIT);   // 48 blocks x 8 waves = 384 waves
    pair_wmma_kernel<<<grid, 256, 0, stream>>>(xyz1, xyz2, W_rad1, G16, ypart);
    reduce_kernel<<<(Z * NTILE * 256) / 256, 256, 0, stream>>>(ypart, mask, s_acc);
    finalize_kernel<<<1, 32, 0, stream>>>(s_acc, W_fc3, W_fc2, out);
}
